// MultiHeadAttention_51376398795564
// MI455X (gfx1250) — compile-verified, matrix operand order checked by reading, not run on hardware
//
#include <hip/hip_runtime.h>
#include <hip/hip_bf16.h>

typedef __attribute__((ext_vector_type(16))) _Float16 v16h;
typedef __attribute__((ext_vector_type(8)))  _Float16 v8h;
typedef __attribute__((ext_vector_type(8)))  float    v8f;

#define D_MODEL   1024
#define NUM_HEADS 16
#define D_K       64
#define SEQ       2048
#define BATCH     2
#define WELEMS    (D_MODEL * D_MODEL)   // 1048576 elems per weight matrix

union Frag {
  v16h v;
  v8h  h[2];
};

static __device__ __forceinline__ v8f zero8() {
  v8f z;
#pragma unroll
  for (int i = 0; i < 8; ++i) z[i] = 0.0f;
  return z;
}

static __device__ __forceinline__ v8h cvt8(const float* p) {
  const float4* q = (const float4*)p;
  float4 a = q[0], b = q[1];
  v8h r;
  r[0] = (_Float16)a.x; r[1] = (_Float16)a.y; r[2] = (_Float16)a.z; r[3] = (_Float16)a.w;
  r[4] = (_Float16)b.x; r[5] = (_Float16)b.y; r[6] = (_Float16)b.z; r[7] = (_Float16)b.w;
  return r;
}

// A/B fragment from a row-major, K-contiguous matrix.
// Layout per CDNA5 ISA: lanes 0-15 hold K {0..7,16..23}, lanes
// 16-31 hold K {8..15,24..31}; row (A) / column (B) index = lane % 16.
static __device__ __forceinline__ Frag load_frag_f32(const float* base, int row0,
                                                     int ld, int k0, int lane) {
  int l16 = lane & 15, hi = lane >> 4;
  const float* p = base + (size_t)(row0 + l16) * ld + k0 + hi * 8;
  Frag f;
  f.h[0] = cvt8(p);
  f.h[1] = cvt8(p + 16);
  return f;
}

static __device__ __forceinline__ Frag load_frag_f16(const _Float16* base, int row0,
                                                     int ld, int k0, int lane) {
  int l16 = lane & 15, hi = lane >> 4;
  const _Float16* p = base + (size_t)(row0 + l16) * ld + k0 + hi * 8;
  Frag f;
  f.h[0] = *(const v8h*)p;
  f.h[1] = *(const v8h*)(p + 16);
  return f;
}

static __device__ __forceinline__ v8f wmma_f16(const Frag& a, const Frag& b, v8f c) {
  return __builtin_amdgcn_wmma_f32_16x16x32_f16(false, a.v, false, b.v,
                                                (short)0, c, false, false);
}

// reductions across the 16-lane half-wave that shares a C-matrix row
static __device__ __forceinline__ float red16_max(float v) {
#pragma unroll
  for (int m = 1; m < 16; m <<= 1) v = fmaxf(v, __shfl_xor(v, m, 32));
  return v;
}
static __device__ __forceinline__ float red16_sum(float v) {
#pragma unroll
  for (int m = 1; m < 16; m <<= 1) v += __shfl_xor(v, m, 32);
  return v;
}

// ---------------------------------------------------------------------------
// Kernel 0: one-shot f32->f16 conversion of the four weight matrices.
// Memory-bound (~24 MB of traffic, ~1 us at 23.3 TB/s) but removes all
// weight-side v_cvt from the GEMM inner loops.
// ---------------------------------------------------------------------------
__global__ __launch_bounds__(256) void cvt_weights_kernel(
    const float* __restrict__ Wq, const float* __restrict__ Wk,
    const float* __restrict__ Wv, const float* __restrict__ Wo,
    _Float16* __restrict__ wh) {
  int gid = blockIdx.x * 256 + threadIdx.x;        // 524288 threads, 8 elems each
  int which = gid >> 17;                           // 131072 threads per matrix
  int idx = (gid & 131071) * 8;
  const float* src = (which == 0) ? Wq : (which == 1) ? Wk : (which == 2) ? Wv : Wo;
  *(v8h*)(wh + (size_t)which * WELEMS + idx) = cvt8(src + idx);
}

// ---------------------------------------------------------------------------
// Kernel 1: fused QKV projection. One wave per 32x64 output tile
// (2 A-frags x 4 B-frags -> 8 WMMAs per 32-wide k-step). Weights are
// pre-converted f16; activations converted on load (amortized 4 WMMAs/frag).
// q scaled by 1/sqrt(d_k); q,k stored [B,H,S,DK] f16; v stored transposed
// [B,H,DK,S] f16 so the P*V B-fragments in flash attention are contiguous.
// ---------------------------------------------------------------------------
__global__ __launch_bounds__(256) void qkv_proj_kernel(
    const float* __restrict__ Q, const float* __restrict__ K, const float* __restrict__ V,
    const _Float16* __restrict__ wh,
    const float* __restrict__ bq, const float* __restrict__ bk, const float* __restrict__ bv,
    _Float16* __restrict__ qh, _Float16* __restrict__ kh, _Float16* __restrict__ vt) {
  int lane = threadIdx.x & 31;
  int wg = blockIdx.x * 8 + (threadIdx.x >> 5);    // 6144 waves
  int which = wg >> 11;                            // 0=q, 1=k, 2=v  (2048 waves each)
  int rem = wg & 2047;
  int mt = rem >> 4, nt = rem & 15;                // 128 m-tiles(32) x 16 n-tiles(64)
  int m0 = mt * 32, n0 = nt * 64;
  const float* X       = (which == 0) ? Q  : (which == 1) ? K  : V;
  const _Float16* W    = wh + (size_t)which * WELEMS;
  const float* bias    = (which == 0) ? bq : (which == 1) ? bk : bv;

  v8f acc0[4], acc1[4];
#pragma unroll
  for (int t = 0; t < 4; ++t) { acc0[t] = zero8(); acc1[t] = zero8(); }

  for (int k0 = 0; k0 < D_MODEL; k0 += 32) {
    Frag a0 = load_frag_f32(X, m0,      D_MODEL, k0, lane);
    Frag a1 = load_frag_f32(X, m0 + 16, D_MODEL, k0, lane);
#pragma unroll
    for (int t = 0; t < 4; ++t) {
      Frag b = load_frag_f16(W, n0 + t * 16, D_MODEL, k0, lane);  // y = x*W^T
      acc0[t] = wmma_f16(a0, b, acc0[t]);
      acc1[t] = wmma_f16(a1, b, acc1[t]);
    }
  }

  int l16 = lane & 15, hi = lane >> 4;
#pragma unroll
  for (int t = 0; t < 4; ++t) {
    int dk = t * 16 + l16;                 // head-local feature (n0 = nt*64)
    float bn = bias[n0 + dk];
#pragma unroll
    for (int mi = 0; mi < 2; ++mi) {
      const v8f& acc = mi ? acc1[t] : acc0[t];
#pragma unroll
      for (int r = 0; r < 8; ++r) {
        int m = m0 + mi * 16 + r + 8 * hi;
        int b = m >> 11, s = m & (SEQ - 1);
        int bh = b * NUM_HEADS + nt;
        float val = acc[r] + bn;
        if (which == 0)
          qh[((size_t)bh * SEQ + s) * D_K + dk] = (_Float16)(val * 0.125f); // 1/sqrt(64)
        else if (which == 1)
          kh[((size_t)bh * SEQ + s) * D_K + dk] = (_Float16)val;
        else
          vt[((size_t)bh * D_K + dk) * SEQ + s] = (_Float16)val;
      }
    }
  }
}

// ---------------------------------------------------------------------------
// Kernel 2: causal flash attention. One wave per (b,h,16-query tile);
// 32 keys per step: QK^T = 4 WMMAs, online softmax in f32, P transposed
// C-layout -> A-layout via per-wave LDS, P*V = 4 WMMAs.
// ---------------------------------------------------------------------------
__global__ __launch_bounds__(256) void flash_attn_kernel(
    const _Float16* __restrict__ qh, const _Float16* __restrict__ kh,
    const _Float16* __restrict__ vt, _Float16* __restrict__ att) {
  __shared__ __align__(16) _Float16 pl[8][16 * 40];  // 40-half row pitch, 16B aligned
  int lane = threadIdx.x & 31;
  int widx = threadIdx.x >> 5;
  int wg = blockIdx.x * 8 + widx;
  int qt = wg & 127, bh = wg >> 7;
  int q0 = qt * 16;
  const _Float16* qb = qh + (size_t)bh * SEQ * D_K;
  const _Float16* kb = kh + (size_t)bh * SEQ * D_K;
  const _Float16* vb = vt + (size_t)bh * D_K * SEQ;
  int l16 = lane & 15, hi = lane >> 4;

  Frag aq0 = load_frag_f16(qb, q0, D_K, 0, lane);
  Frag aq1 = load_frag_f16(qb, q0, D_K, 32, lane);

  v8f o[4];
#pragma unroll
  for (int t = 0; t < 4; ++t) o[t] = zero8();
  float rmax[8], rsum[8];
#pragma unroll
  for (int r = 0; r < 8; ++r) { rmax[r] = -1e30f; rsum[r] = 0.0f; }

  for (int kt = 0; kt < q0 + 16; kt += 32) {
    v8f slo = zero8(), shi = zero8();
    {
      Frag b0 = load_frag_f16(kb, kt, D_K, 0, lane);
      Frag b1 = load_frag_f16(kb, kt, D_K, 32, lane);
      slo = wmma_f16(aq0, b0, slo);
      slo = wmma_f16(aq1, b1, slo);
      Frag b2 = load_frag_f16(kb, kt + 16, D_K, 0, lane);
      Frag b3 = load_frag_f16(kb, kt + 16, D_K, 32, lane);
      shi = wmma_f16(aq0, b2, shi);
      shi = wmma_f16(aq1, b3, shi);
    }
    float alv[8];
#pragma unroll
    for (int r = 0; r < 8; ++r) {
      int row = q0 + r + 8 * hi;                    // global query index
      float sl = (kt + l16 > row)      ? -1e30f : slo[r];
      float sh = (kt + 16 + l16 > row) ? -1e30f : shi[r];
      float mnew = fmaxf(rmax[r], red16_max(fmaxf(sl, sh)));
      float al  = __expf(rmax[r] - mnew);
      float pe0 = __expf(sl - mnew);
      float pe1 = __expf(sh - mnew);
      rsum[r] = rsum[r] * al + red16_sum(pe0 + pe1);
      rmax[r] = mnew;
      alv[r]  = al;
      pl[widx][(r + 8 * hi) * 40 + l16]      = (_Float16)pe0;
      pl[widx][(r + 8 * hi) * 40 + 16 + l16] = (_Float16)pe1;
    }
#pragma unroll
    for (int t = 0; t < 4; ++t)
#pragma unroll
      for (int r = 0; r < 8; ++r) o[t][r] *= alv[r];

    Frag pf;  // re-gather P (16x32) from LDS in A-matrix striping
    const _Float16* pp = &pl[widx][l16 * 40 + hi * 8];
    pf.h[0] = *(const v8h*)pp;
    pf.h[1] = *(const v8h*)(pp + 16);
#pragma unroll
    for (int t = 0; t < 4; ++t) {
      Frag bv = load_frag_f16(vb, t * 16, SEQ, kt, lane);  // V^T rows = d_k
      o[t] = wmma_f16(pf, bv, o[t]);
    }
  }

  int b = bh >> 4, h = bh & 15;
  float inv[8];
#pragma unroll
  for (int r = 0; r < 8; ++r) inv[r] = 1.0f / rsum[r];
#pragma unroll
  for (int t = 0; t < 4; ++t)
#pragma unroll
    for (int r = 0; r < 8; ++r) {
      int s = q0 + r + 8 * hi;
      att[((size_t)(b * SEQ + s)) * D_MODEL + h * D_K + t * 16 + l16] =
          (_Float16)(o[t][r] * inv[r]);
    }
}

// ---------------------------------------------------------------------------
// Kernel 3: output projection y = att * Wo^T + bo, f32 result.
// Both operands f16 in memory -> inner loop is pure b128 loads + WMMA.
// ---------------------------------------------------------------------------
__global__ __launch_bounds__(256) void out_proj_kernel(
    const _Float16* __restrict__ att, const _Float16* __restrict__ woh,
    const float* __restrict__ bo, float* __restrict__ y) {
  int lane = threadIdx.x & 31;
  int wg = blockIdx.x * 8 + (threadIdx.x >> 5);    // 2048 waves
  int mt = wg >> 4, nt = wg & 15;                  // 128 x 16 tiles of 32x64
  int m0 = mt * 32, n0 = nt * 64;

  v8f acc0[4], acc1[4];
#pragma unroll
  for (int t = 0; t < 4; ++t) { acc0[t] = zero8(); acc1[t] = zero8(); }

  for (int k0 = 0; k0 < D_MODEL; k0 += 32) {
    Frag a0 = load_frag_f16(att, m0,      D_MODEL, k0, lane);
    Frag a1 = load_frag_f16(att, m0 + 16, D_MODEL, k0, lane);
#pragma unroll
    for (int t = 0; t < 4; ++t) {
      Frag b = load_frag_f16(woh, n0 + t * 16, D_MODEL, k0, lane);
      acc0[t] = wmma_f16(a0, b, acc0[t]);
      acc1[t] = wmma_f16(a1, b, acc1[t]);
    }
  }

  int l16 = lane & 15, hi = lane >> 4;
#pragma unroll
  for (int t = 0; t < 4; ++t) {
    int n = n0 + t * 16 + l16;
    float bn = bo[n];
#pragma unroll
    for (int mi = 0; mi < 2; ++mi) {
      const v8f& acc = mi ? acc1[t] : acc0[t];
#pragma unroll
      for (int r = 0; r < 8; ++r) {
        int m = m0 + mi * 16 + r + 8 * hi;
        y[(size_t)m * D_MODEL + n] = acc[r] + bn;
      }
    }
  }
}

extern "C" void kernel_launch(void* const* d_in, const int* in_sizes, int n_in,
                              void* d_out, int out_size, void* d_ws, size_t ws_size,
                              hipStream_t stream) {
  const float* Q  = (const float*)d_in[0];
  const float* K  = (const float*)d_in[1];
  const float* V  = (const float*)d_in[2];
  // d_in[3] is the causal tril mask; applied analytically in-kernel.
  const float* Wq = (const float*)d_in[4];
  const float* bq = (const float*)d_in[5];
  const float* Wk = (const float*)d_in[6];
  const float* bk = (const float*)d_in[7];
  const float* Wv = (const float*)d_in[8];
  const float* bv = (const float*)d_in[9];
  const float* Wo = (const float*)d_in[10];
  const float* bo = (const float*)d_in[11];
  float* y = (float*)d_out;

  char* ws = (char*)d_ws;
  const size_t HBUF = (size_t)BATCH * NUM_HEADS * SEQ * D_K * sizeof(_Float16); // 8 MB
  _Float16* qh  = (_Float16*)(ws);                 //  8 MB
  _Float16* kh  = (_Float16*)(ws + HBUF);          //  8 MB
  _Float16* vt  = (_Float16*)(ws + 2 * HBUF);      //  8 MB
  _Float16* att = (_Float16*)(ws + 3 * HBUF);      //  8 MB
  _Float16* wh  = (_Float16*)(ws + 4 * HBUF);      //  8 MB (4 weight matrices f16)

  // 4 * 1M elems / (256 thr * 8 elems) = 2048 blocks
  cvt_weights_kernel<<<2048, 256, 0, stream>>>(Wq, Wk, Wv, Wo, wh);
  // 3 weights * 128 m-tiles(32) * 16 n64-tiles = 6144 waves / 8 per block
  qkv_proj_kernel<<<768, 256, 0, stream>>>(Q, K, V, wh, bq, bk, bv, qh, kh, vt);
  // 32 (b,h) * 128 q-tiles = 4096 waves
  flash_attn_kernel<<<512, 256, 0, stream>>>(qh, kh, vt, att);
  // 128 m-tiles(32) * 16 n64-tiles = 2048 waves
  out_proj_kernel<<<256, 256, 0, stream>>>(att, wh + (size_t)3 * WELEMS, bo, y);
}
